// LinkPredictor_6622839570447
// MI455X (gfx1250) — compile-verified, matrix operand order checked by reading, not run on hardware
//
#include <hip/hip_runtime.h>
#include <hip/hip_bf16.h>

typedef __attribute__((ext_vector_type(16))) __bf16         v16bf;
typedef __attribute__((ext_vector_type(16))) unsigned short v16u;
typedef __attribute__((ext_vector_type(8)))  float          v8f;
typedef __attribute__((ext_vector_type(8)))  unsigned int   v8u;

// ---------------------------------------------------------------- helpers
__device__ inline unsigned short f2bf(float f) {        // native v_cvt f32->bf16
    union { __bf16 b; unsigned short u; } c;
    c.b = (__bf16)f;
    return c.u;
}
__device__ inline void atomAddG(float* p, float v) {    // global_atomic_add_f32
    __hip_atomic_fetch_add(p, v, __ATOMIC_RELAXED, __HIP_MEMORY_SCOPE_AGENT);
}
__device__ inline void atomAddS(float* p, float v) {    // ds_add_f32
    __hip_atomic_fetch_add(p, v, __ATOMIC_RELAXED, __HIP_MEMORY_SCOPE_WORKGROUP);
}

// ---------------------------------------------------------------- GEMM (bf16 WMMA, f32 accum)
// C[M x N] = A[M x 128] @ W[128 x N]  (+ bias) (+= existing C if accFlag)
// Block: 256 threads = 8 waves; block tile = 128 rows x N.
// W staged in LDS pre-swizzled into per-lane WMMA B-fragments (2x ds_load_b128 each).
// All A raw data (4 k-steps) prefetched up front so global loads overlap the WMMA pipe.
template <int NT>   // NT = N/16 (8 or 4)
__global__ __launch_bounds__(256) void k_gemm(const float* __restrict__ A,
                                              const float* __restrict__ W,
                                              const float* __restrict__ bias,
                                              float* __restrict__ C,
                                              int M, int accFlag) {
    constexpr int K       = 128;
    constexpr int N       = NT * 16;
    constexpr int KSTEPS  = K / 32;          // 4
    constexpr int FRAGS   = KSTEPS * NT;
    constexpr int LSTRIDE = 24;              // ushorts per lane (16 used + pad, 48B)
    constexpr int FSTRIDE = 32 * LSTRIDE;    // ushorts per fragment
    __shared__ __align__(16) unsigned short sWf[FRAGS * FSTRIDE];

    const int tid = threadIdx.x;

    // stage + swizzle W -> bf16 fragments
    for (int i = tid; i < FRAGS * 512; i += 256) {
        const int e  = i & 15;
        const int ln = (i >> 4) & 31;
        const int f  = i >> 9;               // fragment id = ks*NT + t
        const int t  = f % NT;
        const int ks = f / NT;
        const int col = t * 16 + (ln & 15);
        const int kk  = ks * 32 + (ln >> 4) * 8 + (e & 7) + ((e >> 3) << 4);
        sWf[f * FSTRIDE + ln * LSTRIDE + e] = f2bf(W[kk * N + col]);
    }
    __syncthreads();

    const int w    = tid >> 5;
    const int lane = tid & 31;
    const int half = lane >> 4;
    const int l15  = lane & 15;

    const int rowBase = blockIdx.x * 128 + w * 16;
    const int row     = rowBase + l15;
    const int rowC    = (row < M) ? row : (M - 1);   // clamp loads; stores guarded

    v8f acc[NT];
#pragma unroll
    for (int t = 0; t < NT; ++t)
#pragma unroll
        for (int j = 0; j < 8; ++j) acc[t][j] = 0.0f;

    // ---- prefetch all A raw data (16-bit A layout: lane=row l15,
    //      half 0 -> K = k0+0..7 and k0+16..23; half 1 -> +8) ----
    const float* ap = A + (size_t)rowC * K;
    float4 ar[KSTEPS][4];
#pragma unroll
    for (int ks = 0; ks < KSTEPS; ++ks) {
        const float* p = ap + ks * 32 + half * 8;
        ar[ks][0] = *(const float4*)(p);
        ar[ks][1] = *(const float4*)(p + 4);
        ar[ks][2] = *(const float4*)(p + 16);
        ar[ks][3] = *(const float4*)(p + 20);
    }

#pragma unroll
    for (int ks = 0; ks < KSTEPS; ++ks) {
        v16u au;
#pragma unroll
        for (int q = 0; q < 4; ++q) {
            au[q * 4 + 0] = f2bf(ar[ks][q].x);
            au[q * 4 + 1] = f2bf(ar[ks][q].y);
            au[q * 4 + 2] = f2bf(ar[ks][q].z);
            au[q * 4 + 3] = f2bf(ar[ks][q].w);
        }
        v16bf afrag = __builtin_bit_cast(v16bf, au);

#pragma unroll
        for (int t = 0; t < NT; ++t) {
            const v8u bq = *(const v8u*)(&sWf[(ks * NT + t) * FSTRIDE + lane * LSTRIDE]);
            v16bf bfrag = __builtin_bit_cast(v16bf, bq);
            acc[t] = __builtin_amdgcn_wmma_f32_16x16x32_bf16(
                         false, afrag, false, bfrag, (short)0, acc[t], false, false);
        }
    }

    // epilogue: C/D layout — VGPR v: row = v + 8*half, col = l15 (per 16x16 tile)
#pragma unroll
    for (int t = 0; t < NT; ++t) {
        const int col = t * 16 + l15;
#pragma unroll
        for (int v = 0; v < 8; ++v) {
            const int r = rowBase + half * 8 + v;
            if (r < M) {
                float val = acc[t][v];
                if (accFlag) val += C[(size_t)r * N + col];
                if (bias)    val += bias[col];
                C[(size_t)r * N + col] = val;
            }
        }
    }
}

// ---------------------------------------------------------------- edge kernels
__global__ void k_deg(const int* __restrict__ dst, float* __restrict__ deg, int E) {
    int i = blockIdx.x * blockDim.x + threadIdx.x;
    int n = gridDim.x * blockDim.x;
    for (; i < E; i += n) atomAddG(&deg[dst[i]], 1.0f);
}

__global__ void k_dinv(float* __restrict__ d, int n) {
    int i = blockIdx.x * blockDim.x + threadIdx.x;
    if (i < n) d[i] = rsqrtf(d[i] + 1.0f);
}

// wave-per-edge scatter-add: out[dst] += feat[src] * norm   (F = 128 or 64)
__global__ __launch_bounds__(256) void k_scatter(const float* __restrict__ feat,
                                                 float* __restrict__ out,
                                                 const int* __restrict__ sidx,
                                                 const int* __restrict__ didx,
                                                 const float* __restrict__ dinv,
                                                 int E, int F) {
    const int lane = threadIdx.x & 31;
    int wid = (blockIdx.x * blockDim.x + threadIdx.x) >> 5;
    const int nw = (gridDim.x * blockDim.x) >> 5;
    if (F == 128) {
        for (int e = wid; e < E; e += nw) {
            const int s = sidx[e], d = didx[e];
            const float nm = dinv ? dinv[s] * dinv[d] : 1.0f;
            const float4 v = *(const float4*)(feat + (size_t)s * 128 + lane * 4);
            float* op = out + (size_t)d * 128 + lane * 4;
            atomAddG(op + 0, v.x * nm); atomAddG(op + 1, v.y * nm);
            atomAddG(op + 2, v.z * nm); atomAddG(op + 3, v.w * nm);
        }
    } else {  // F == 64
        for (int e = wid; e < E; e += nw) {
            const int s = sidx[e], d = didx[e];
            const float nm = dinv ? dinv[s] * dinv[d] : 1.0f;
            const float2 v = *(const float2*)(feat + (size_t)s * 64 + lane * 2);
            float* op = out + (size_t)d * 64 + lane * 2;
            atomAddG(op + 0, v.x * nm); atomAddG(op + 1, v.y * nm);
        }
    }
}

// paper->label scatter (128 dst rows): LDS accumulation (ds_add_f32) + one global flush
__global__ __launch_bounds__(256) void k_scatter_label(const float* __restrict__ feat,
                                                       float* __restrict__ out,
                                                       const int* __restrict__ sidx,
                                                       const int* __restrict__ didx,
                                                       int E, int NL) {
    __shared__ float acc[128 * 128];   // 64 KB
    for (int i = threadIdx.x; i < NL * 128; i += 256) acc[i] = 0.0f;
    __syncthreads();
    const int lane = threadIdx.x & 31;
    int wid = (blockIdx.x * blockDim.x + threadIdx.x) >> 5;
    const int nw = (gridDim.x * blockDim.x) >> 5;
    for (int e = wid; e < E; e += nw) {
        const int s = sidx[e], d = didx[e];
        const float4 v = *(const float4*)(feat + (size_t)s * 128 + lane * 4);
        float* ap = acc + d * 128 + lane * 4;
        atomAddS(ap + 0, v.x); atomAddS(ap + 1, v.y);
        atomAddS(ap + 2, v.z); atomAddS(ap + 3, v.w);
    }
    __syncthreads();
    for (int i = threadIdx.x; i < NL * 128; i += 256) atomAddG(&out[i], acc[i]);
}

// out = [relu] 0.5 * ( scat + dinv^2 * hg + gcn_b + orev )
__global__ void k_combine(const float* __restrict__ scat, const float* __restrict__ hg,
                          const float* __restrict__ dinv, const float* __restrict__ gb,
                          const float* __restrict__ orev, float* __restrict__ out,
                          int M, int fshift, int doRelu) {
    const int F = 1 << fshift;
    size_t i = (size_t)blockIdx.x * blockDim.x + threadIdx.x;
    const size_t n = (size_t)M << fshift;
    const size_t stride = (size_t)gridDim.x * blockDim.x;
    for (; i < n; i += stride) {
        const int r = (int)(i >> fshift), c = (int)(i & (F - 1));
        const float di = dinv[r];
        float v = 0.5f * ((scat[i] + di * di * hg[i] + gb[c]) + orev[i]);
        out[i] = doRelu ? fmaxf(v, 0.0f) : v;
    }
}

__global__ void k_relu(const float* __restrict__ in, float* __restrict__ out, int n) {
    int i = blockIdx.x * blockDim.x + threadIdx.x;
    if (i < n) out[i] = fmaxf(in[i], 0.0f);
}

// ---------------------------------------------------------------- host
extern "C" void kernel_launch(void* const* d_in, const int* in_sizes, int n_in,
                              void* d_out, int out_size, void* d_ws, size_t ws_size,
                              hipStream_t stream) {
    const float* xp = (const float*)d_in[0];
    const float* xl = (const float*)d_in[1];
    const int* cs   = (const int*)d_in[2];
    const int* cd   = (const int*)d_in[3];
    const int* isrc = (const int*)d_in[4];
    const int* isd  = (const int*)d_in[5];
    const float* l1_gcnW    = (const float*)d_in[6];
    const float* l1_gcnb    = (const float*)d_in[7];
    const float* l1_isRelW  = (const float*)d_in[8];
    const float* l1_isRelb  = (const float*)d_in[9];
    const float* l1_isRootW = (const float*)d_in[10];
    const float* l1_revRelW = (const float*)d_in[11];
    const float* l1_revRelb = (const float*)d_in[12];
    const float* l1_revRootW= (const float*)d_in[13];
    const float* l2_gcnW    = (const float*)d_in[14];
    const float* l2_gcnb    = (const float*)d_in[15];
    const float* l2_isRelW  = (const float*)d_in[16];
    const float* l2_isRelb  = (const float*)d_in[17];
    const float* l2_isRootW = (const float*)d_in[18];
    const float* l2_revRelW = (const float*)d_in[19];
    const float* l2_revRelb = (const float*)d_in[20];
    const float* l2_revRootW= (const float*)d_in[21];

    const int NP  = in_sizes[0] / 128;   // 100000
    const int NL  = in_sizes[1] / 128;   // 128
    const int EC  = in_sizes[2];         // 1.6M
    const int EI  = in_sizes[4];         // 100K
    const int EMB = in_sizes[14] / 128;  // 64

    // workspace carve
    float* ws   = (float*)d_ws;
    size_t off  = 0;
    float* dinv = ws + off; off += (size_t)NP;
    float* A    = ws + off; off += (size_t)NP * 128;   // h_gcn1 -> agg_rev2
    float* Bb   = ws + off; off += (size_t)NP * 128;   // scat1 -> h_gcn2 | o_rev2
    float* Cc   = ws + off; off += (size_t)NP * 128;   // agg_rev1 -> hp
    float* Dd   = ws + off; off += (size_t)NP * 128;   // o_rev1 -> scat2
    float* aggL = ws + off; off += (size_t)NL * 128;
    float* oL   = ws + off; off += (size_t)NL * 128;
    float* hl   = ws + off; off += (size_t)NL * 128;

    float* zp = (float*)d_out;
    float* zl = (float*)d_out + (size_t)NP * EMB;

    const int gBig   = 4096;                       // elementwise
    const int gEdges = 8192;                       // cites scatter
    const int gSmall = 2048;
    const int gGemmP = (NP + 127) / 128;
    const int gGemmL = (NL + 127) / 128;

    // degrees (shared by both layers)
    hipMemsetAsync(dinv, 0, (size_t)NP * 4, stream);
    k_deg<<<gSmall, 256, 0, stream>>>(cd, dinv, EC);
    k_dinv<<<(NP + 255) / 256, 256, 0, stream>>>(dinv, NP);

    // ---------------- layer 1 (128 -> 128) ----------------
    k_gemm<8><<<gGemmP, 256, 0, stream>>>(xp, l1_gcnW, nullptr, A, NP, 0);        // h_gcn1
    hipMemsetAsync(Bb, 0, (size_t)NP * 128 * 4, stream);
    k_scatter<<<gEdges, 256, 0, stream>>>(A, Bb, cs, cd, dinv, EC, 128);          // cites agg
    hipMemsetAsync(Cc, 0, (size_t)NP * 128 * 4, stream);
    k_scatter<<<gSmall, 256, 0, stream>>>(xl, Cc, isd, isrc, nullptr, EI, 128);   // label->paper agg
    k_gemm<8><<<gGemmP, 256, 0, stream>>>(Cc, l1_revRelW, l1_revRelb, Dd, NP, 0); // o_rev1
    k_gemm<8><<<gGemmP, 256, 0, stream>>>(xp, l1_revRootW, nullptr, Dd, NP, 1);   // += root
    k_combine<<<gBig, 256, 0, stream>>>(Bb, A, dinv, l1_gcnb, Dd, Cc, NP, 7, 1);  // hp -> Cc

    hipMemsetAsync(aggL, 0, (size_t)NL * 128 * 4, stream);
    k_scatter_label<<<128, 256, 0, stream>>>(xp, aggL, isrc, isd, EI, NL);        // paper->label agg
    k_gemm<8><<<gGemmL, 256, 0, stream>>>(aggL, l1_isRelW, l1_isRelb, oL, NL, 0);
    k_gemm<8><<<gGemmL, 256, 0, stream>>>(xl, l1_isRootW, nullptr, oL, NL, 1);
    k_relu<<<(NL * 128 + 255) / 256, 256, 0, stream>>>(oL, hl, NL * 128);         // hl

    // ---------------- layer 2 (128 -> 64) ----------------
    float* hgcn2 = Bb;                       // NP x 64
    float* orev2 = Bb + (size_t)NP * EMB;    // NP x 64
    k_gemm<4><<<gGemmP, 256, 0, stream>>>(Cc, l2_gcnW, nullptr, hgcn2, NP, 0);    // h_gcn2
    hipMemsetAsync(Dd, 0, (size_t)NP * EMB * 4, stream);
    k_scatter<<<gEdges, 256, 0, stream>>>(hgcn2, Dd, cs, cd, dinv, EC, EMB);      // cites agg
    hipMemsetAsync(A, 0, (size_t)NP * 128 * 4, stream);
    k_scatter<<<gSmall, 256, 0, stream>>>(hl, A, isd, isrc, nullptr, EI, 128);    // label->paper agg
    k_gemm<4><<<gGemmP, 256, 0, stream>>>(A, l2_revRelW, l2_revRelb, orev2, NP, 0);
    k_gemm<4><<<gGemmP, 256, 0, stream>>>(Cc, l2_revRootW, nullptr, orev2, NP, 1);
    k_combine<<<gBig, 256, 0, stream>>>(Dd, hgcn2, dinv, l2_gcnb, orev2, zp, NP, 6, 0);

    hipMemsetAsync(aggL, 0, (size_t)NL * 128 * 4, stream);
    k_scatter_label<<<128, 256, 0, stream>>>(Cc, aggL, isrc, isd, EI, NL);
    k_gemm<4><<<gGemmL, 256, 0, stream>>>(aggL, l2_isRelW, l2_isRelb, zl, NL, 0);
    k_gemm<4><<<gGemmL, 256, 0, stream>>>(hl, l2_isRootW, nullptr, zl, NL, 1);
}
